// RevGAT_2216203124986
// MI455X (gfx1250) — compile-verified, matrix operand order checked by reading, not run on hardware
//
#include <hip/hip_runtime.h>
#include <hip/hip_bf16.h>
#include <math.h>

// ---------------- problem constants (match reference) ----------------
#define N_NODES   50000
#define N_EDGES   400000
#define IN_FEATS  256
#define HID       64
#define HEADS     4
#define CLASSES   40
#define WFULL     256          // HEADS*HID
#define GW        128          // WFULL / GROUP
#define NEG_SLOPE 0.2f
#define EPS_BN    1e-5f

typedef __attribute__((ext_vector_type(16))) _Float16 v16h;
typedef __attribute__((ext_vector_type(8)))  float    v8f;

// =====================================================================
// Utility kernels
// =====================================================================
__global__ void fill_kernel(float* __restrict__ p, long long n, float v) {
  long long i = (long long)blockIdx.x * blockDim.x + threadIdx.x;
  long long stride = (long long)gridDim.x * blockDim.x;
  for (; i < n; i += stride) p[i] = v;
}

__global__ void degree_kernel(const int* __restrict__ src, const int* __restrict__ dst,
                              float* __restrict__ dout, float* __restrict__ din, int E) {
  int i = blockIdx.x * blockDim.x + threadIdx.x;
  if (i < E) {
    atomicAdd(&dout[src[i]], 1.0f);
    atomicAdd(&din[dst[i]], 1.0f);
  }
}

__global__ void degnorm_kernel(float* __restrict__ dout, float* __restrict__ din, int n) {
  int i = blockIdx.x * blockDim.x + threadIdx.x;
  if (i < n) {
    dout[i] = rsqrtf(fmaxf(dout[i], 1.0f));  // out_isqrt
    din[i]  = sqrtf (fmaxf(din[i],  1.0f));  // in_sqrt
  }
}

// =====================================================================
// Dense GEMM via v_wmma_f32_16x16x32_f16:  C[M,Nout] = A[M,K] * W[Nout,K]^T
// One wave computes a 16(M) x 64(N) strip: the A fragment (and its
// f32->f16 conversion) is reused across 4 WMMAs per K-step, so per-WMMA
// activation traffic drops 4x; B rows are weights resident in L2.
// A-layout (ISA 7.12.2, 16-bit A 16x32): lane L -> row M=L%16,
//   elems 0..7 = K[h0..h0+7], elems 8..15 = K[16+h0..16+h0+7], h0=(L<16?0:8).
// B-layout mirrors A with N=L%16 (W row-major in K, so B(k,n)=W[n][k]).
// =====================================================================
__global__ __launch_bounds__(256)
void gemm_wmma_kernel(const float* __restrict__ A, const float* __restrict__ Wt,
                      float* __restrict__ C, int M, int K, int Nout,
                      int tilesN64, int totalTiles) {
  int wavesPerBlock = blockDim.x >> 5;
  int tile = blockIdx.x * wavesPerBlock + (threadIdx.x >> 5);
  if (tile >= totalTiles) return;               // wave-uniform
  int lane = threadIdx.x & 31;
  int tm = tile / tilesN64;
  int tn = tile - tm * tilesN64;                // 64-column strip index
  int half = lane >> 4;                         // 0 or 1
  int h0 = half * 8;
  int mrow = tm * 16 + (lane & 15);
  const float* arow = A + (size_t)mrow * K;

  int ncol[4];
  const float* brow[4];
#pragma unroll
  for (int t = 0; t < 4; ++t) {
    ncol[t] = tn * 64 + 16 * t + (lane & 15);
    int nc = (ncol[t] < Nout) ? ncol[t] : (Nout - 1);
    brow[t] = Wt + (size_t)nc * K;
  }

  v8f acc0 = {}, acc1 = {}, acc2 = {}, acc3 = {};
  for (int kk = 0; kk < K; kk += 32) {
    __builtin_prefetch(arow + kk + 128, 0, 1);   // stream next A cachelines
    v16h a;
#pragma unroll
    for (int j = 0; j < 8; ++j) {
      a[j]     = (_Float16)arow[kk + h0 + j];
      a[8 + j] = (_Float16)arow[kk + 16 + h0 + j];
    }
    v16h b0, b1, b2, b3;
#pragma unroll
    for (int j = 0; j < 8; ++j) {
      b0[j]     = (_Float16)brow[0][kk + h0 + j];
      b0[8 + j] = (_Float16)brow[0][kk + 16 + h0 + j];
      b1[j]     = (_Float16)brow[1][kk + h0 + j];
      b1[8 + j] = (_Float16)brow[1][kk + 16 + h0 + j];
      b2[j]     = (_Float16)brow[2][kk + h0 + j];
      b2[8 + j] = (_Float16)brow[2][kk + 16 + h0 + j];
      b3[j]     = (_Float16)brow[3][kk + h0 + j];
      b3[8 + j] = (_Float16)brow[3][kk + 16 + h0 + j];
    }
    acc0 = __builtin_amdgcn_wmma_f32_16x16x32_f16(false, a, false, b0, (short)0, acc0, false, false);
    acc1 = __builtin_amdgcn_wmma_f32_16x16x32_f16(false, a, false, b1, (short)0, acc1, false, false);
    acc2 = __builtin_amdgcn_wmma_f32_16x16x32_f16(false, a, false, b2, (short)0, acc2, false, false);
    acc3 = __builtin_amdgcn_wmma_f32_16x16x32_f16(false, a, false, b3, (short)0, acc3, false, false);
  }

  // store: acc[v] -> row tm*16 + v + 8*half, col ncol[t]  (mask only at store)
  v8f* accs[4] = { &acc0, &acc1, &acc2, &acc3 };
#pragma unroll
  for (int t = 0; t < 4; ++t) {
    if (ncol[t] < Nout) {
#pragma unroll
      for (int v = 0; v < 8; ++v) {
        int r = tm * 16 + v + half * 8;
        C[(size_t)r * Nout + ncol[t]] = (*accs[t])[v];
      }
    }
  }
}

// =====================================================================
// GAT node-side attention terms.
//   el[n,h] = out_isqrt[n] * sum_d ft0[n,h,d]*attn_l[h,d]   (normalized src feat)
//   er[n,h] =               sum_d ft0[n,h,d]*attn_r[h,d]   (pre-norm dst feat)
// then ft <- ft0 * out_isqrt (feat_src used in messages).
// =====================================================================
__global__ void node_attn_kernel(float* __restrict__ ft,
                                 const float* __restrict__ al, const float* __restrict__ ar,
                                 const float* __restrict__ out_isqrt,
                                 float* __restrict__ el, float* __restrict__ er,
                                 int n, int H, int D) {
  int i = blockIdx.x * blockDim.x + threadIdx.x;
  if (i >= n * H) return;
  int node = i / H, h = i - node * H;
  float s = out_isqrt[node];
  float* f = ft + (size_t)node * H * D + (size_t)h * D;
  const float* alh = al + h * D;
  const float* arh = ar + h * D;
  float sl = 0.f, sr = 0.f;
  for (int d = 0; d < D; ++d) { float v = f[d]; sl += v * alh[d]; sr += v * arh[d]; }
  el[i] = sl * s;
  er[i] = sr;
  for (int d = 0; d < D; ++d) f[d] *= s;
}

__device__ __forceinline__ float leaky(float v) {
  return (v >= 0.f) ? v : NEG_SLOPE * v;
}

__device__ __forceinline__ void atomicMaxFloat(float* addr, float val) {
  int* ia = reinterpret_cast<int*>(addr);
  int old = __float_as_int(*addr);
  while (__int_as_float(old) < val) {
    int assumed = old;
    old = atomicCAS(ia, assumed, __float_as_int(val));
    if (old == assumed) break;
  }
}

__global__ void edge_max_kernel(const int* __restrict__ src, const int* __restrict__ dst,
                                const float* __restrict__ el, const float* __restrict__ er,
                                float* __restrict__ emax, int E, int H) {
  int i = blockIdx.x * blockDim.x + threadIdx.x;
  if (i >= E * H) return;
  int e = i / H, h = i - e * H;
  float v = leaky(el[src[e] * H + h] + er[dst[e] * H + h]);
  atomicMaxFloat(&emax[dst[e] * H + h], v);
}

__global__ void edge_exp_kernel(const int* __restrict__ src, const int* __restrict__ dst,
                                const float* __restrict__ el, const float* __restrict__ er,
                                const float* __restrict__ emax,
                                float* __restrict__ ex, float* __restrict__ denom,
                                int E, int H) {
  int i = blockIdx.x * blockDim.x + threadIdx.x;
  if (i >= E * H) return;
  int e = i / H, h = i - e * H;
  int d = dst[e];
  float m = emax[d * H + h];
  if (!(m >= -3.0e38f)) m = 0.0f;   // mirrors jnp.where(isfinite(emax), emax, 0)
  float v = leaky(el[src[e] * H + h] + er[d * H + h]);
  float x = __expf(v - m);
  ex[i] = x;
  atomicAdd(&denom[d * H + h], x);
}

// agg[dst,h,d] += ft[src,h,d] * ex/max(denom,1e-9)
__global__ void message_kernel(const int* __restrict__ src, const int* __restrict__ dst,
                               const float* __restrict__ ft,
                               const float* __restrict__ ex, const float* __restrict__ denom,
                               float* __restrict__ agg, int E, int H, int D) {
  long long total = (long long)E * H * D;
  long long i = (long long)blockIdx.x * blockDim.x + threadIdx.x;
  long long stride = (long long)gridDim.x * blockDim.x;
  int HD = H * D;
  for (; i < total; i += stride) {
    int e = (int)(i / HD);
    int r = (int)(i - (long long)e * HD);
    int h = r / D;
    int sN = src[e], dN = dst[e];
    float a = ex[(size_t)e * H + h] / fmaxf(denom[(size_t)dN * H + h], 1e-9f);
    atomicAdd(&agg[(size_t)dN * HD + r], ft[(size_t)sN * HD + r] * a);
  }
}

// out[n, ooff+c] (stride ostride) = agg*in_sqrt + res  (+ self) (+ bias)
__global__ void finalize_kernel(const float* __restrict__ agg, const float* __restrict__ res,
                                const float* __restrict__ in_sqrt, float* __restrict__ out,
                                int n, int HD, int ostride, int ooff,
                                int addSelf, const float* __restrict__ bias) {
  int i = blockIdx.x * blockDim.x + threadIdx.x;
  if (i >= n * HD) return;
  int node = i / HD, c = i - node * HD;
  float v = agg[i] * in_sqrt[node] + res[i];
  float* o = out + (size_t)node * ostride + ooff + c;
  if (addSelf) v += *o;
  if (bias) v += bias[c];
  *o = v;
}

// =====================================================================
// BatchNorm (training stats, biased var) + ReLU
// =====================================================================
__global__ void bn_stats_kernel(const float* __restrict__ x, int n, int C, int ld, int off,
                                float* __restrict__ sum, float* __restrict__ sumsq) {
  int c = blockIdx.x * blockDim.x + threadIdx.x;
  if (c >= C) return;
  int chunks = gridDim.y;
  int rows = (n + chunks - 1) / chunks;
  int r0 = blockIdx.y * rows;
  int r1 = r0 + rows; if (r1 > n) r1 = n;
  float s = 0.f, s2 = 0.f;
  for (int r = r0; r < r1; ++r) {
    float v = x[(size_t)r * ld + off + c];
    s += v; s2 += v * v;
  }
  atomicAdd(&sum[c], s);
  atomicAdd(&sumsq[c], s2);
}

__global__ void bn_moments_kernel(float* __restrict__ sum, float* __restrict__ sumsq,
                                  int n, int C) {
  int c = blockIdx.x * blockDim.x + threadIdx.x;
  if (c >= C) return;
  float inv_n = 1.0f / (float)n;
  float m = sum[c] * inv_n;
  float var = sumsq[c] * inv_n - m * m;
  sum[c] = m;                       // mean
  sumsq[c] = rsqrtf(var + EPS_BN);  // 1/sqrt(var+eps)
}

__global__ void bn_apply_kernel(const float* __restrict__ x,
                                const float* __restrict__ mean, const float* __restrict__ rstd,
                                const float* __restrict__ gamma, const float* __restrict__ beta,
                                float* __restrict__ z, int n, int C, int ld, int off) {
  int i = blockIdx.x * blockDim.x + threadIdx.x;
  if (i >= n * C) return;
  int node = i / C, c = i - node * C;
  float v = gamma[c] * (x[(size_t)node * ld + off + c] - mean[c]) * rstd[c] + beta[c];
  z[i] = fmaxf(v, 0.0f);
}

// =====================================================================
// Host-side helpers (launch sequences on `stream`; no allocation)
// =====================================================================
namespace {

struct Ws {
  float *out_isqrt, *in_sqrt, *h, *ft, *res, *agg, *z;
  float *el, *er, *emax, *denom, *ex, *bnsum;  // bnsum: 512 floats (sum|sumsq)
};

inline int cdiv(int a, int b) { return (a + b - 1) / b; }

void gat_conv(hipStream_t s, const Ws& w, const float* x, int F, int H, int D,
              const float* Wp, const float* al, const float* ar, const float* resWp,
              const int* src, const int* dst,
              float* out, int ostride, int ooff, int addSelf, const float* bias) {
  const int n = N_NODES, E = N_EDGES;
  const int HD = H * D;
  const int tilesN64 = cdiv(HD, 64);
  const int totalTiles = (n / 16) * tilesN64;     // n = 50000 = 3125*16
  const int gemmBlocks = cdiv(totalTiles, 8);      // 8 waves / 256-thread block

  gemm_wmma_kernel<<<gemmBlocks, 256, 0, s>>>(x, Wp,    w.ft,  n, F, HD, tilesN64, totalTiles);
  gemm_wmma_kernel<<<gemmBlocks, 256, 0, s>>>(x, resWp, w.res, n, F, HD, tilesN64, totalTiles);

  node_attn_kernel<<<cdiv(n * H, 256), 256, 0, s>>>(w.ft, al, ar, w.out_isqrt,
                                                    w.el, w.er, n, H, D);

  fill_kernel<<<256, 256, 0, s>>>(w.emax,  (long long)n * H, -INFINITY);
  fill_kernel<<<256, 256, 0, s>>>(w.denom, (long long)n * H, 0.0f);
  fill_kernel<<<2048, 256, 0, s>>>(w.agg,  (long long)n * HD, 0.0f);

  edge_max_kernel<<<cdiv(E * H, 256), 256, 0, s>>>(src, dst, w.el, w.er, w.emax, E, H);
  edge_exp_kernel<<<cdiv(E * H, 256), 256, 0, s>>>(src, dst, w.el, w.er, w.emax,
                                                   w.ex, w.denom, E, H);
  message_kernel<<<8192, 256, 0, s>>>(src, dst, w.ft, w.ex, w.denom, w.agg, E, H, D);
  finalize_kernel<<<cdiv(n * HD, 256), 256, 0, s>>>(w.agg, w.res, w.in_sqrt, out,
                                                    n, HD, ostride, ooff, addSelf, bias);
}

void bn_relu(hipStream_t s, const Ws& w, const float* x, int ld, int off, int C,
             const float* gamma, const float* beta, float* z) {
  const int n = N_NODES;
  float* sum = w.bnsum;
  float* sumsq = w.bnsum + 256;
  fill_kernel<<<2, 256, 0, s>>>(w.bnsum, 512, 0.0f);
  dim3 g(cdiv(C, 64), 96);
  bn_stats_kernel<<<g, 64, 0, s>>>(x, n, C, ld, off, sum, sumsq);
  bn_moments_kernel<<<cdiv(C, 64), 64, 0, s>>>(sum, sumsq, n, C);
  bn_apply_kernel<<<cdiv(n * C, 256), 256, 0, s>>>(x, sum, sumsq, gamma, beta,
                                                   z, n, C, ld, off);
}

} // namespace

// =====================================================================
// Entry point
// =====================================================================
extern "C" void kernel_launch(void* const* d_in, const int* in_sizes, int n_in,
                              void* d_out, int out_size, void* d_ws, size_t ws_size,
                              hipStream_t stream) {
  (void)in_sizes; (void)n_in; (void)out_size; (void)ws_size;
  const float* x         = (const float*)d_in[0];
  const int*   src       = (const int*)  d_in[1];
  const int*   dst       = (const int*)  d_in[2];
  const float* W0        = (const float*)d_in[3];
  const float* al0       = (const float*)d_in[4];
  const float* ar0       = (const float*)d_in[5];
  const float* resW0     = (const float*)d_in[6];
  const float* bn_gamma  = (const float*)d_in[7];   // (2,2,128)
  const float* bn_beta   = (const float*)d_in[8];
  const float* mid_W     = (const float*)d_in[9];   // (2,2,128,128)
  const float* mid_al    = (const float*)d_in[10];  // (2,2,4,32)
  const float* mid_ar    = (const float*)d_in[11];
  const float* mid_resW  = (const float*)d_in[12];
  const float* norm_g    = (const float*)d_in[13];
  const float* norm_b    = (const float*)d_in[14];
  const float* W_last    = (const float*)d_in[15];  // (40,256)
  const float* al_last   = (const float*)d_in[16];  // (1,40)
  const float* ar_last   = (const float*)d_in[17];
  const float* resW_last = (const float*)d_in[18];
  const float* bias_last = (const float*)d_in[19];
  float* out = (float*)d_out;

  // ---- carve workspace (floats) ----
  float* f = (float*)d_ws;
  size_t o = 0;
  Ws w;
  const size_t N = N_NODES, E = N_EDGES;
  w.out_isqrt = f + o; o += N;
  w.in_sqrt   = f + o; o += N;
  w.h         = f + o; o += N * WFULL;
  w.ft        = f + o; o += N * WFULL;
  w.res       = f + o; o += N * WFULL;
  w.agg       = f + o; o += N * WFULL;
  w.z         = f + o; o += N * WFULL;
  w.el        = f + o; o += N * HEADS;
  w.er        = f + o; o += N * HEADS;
  w.emax      = f + o; o += N * HEADS;
  w.denom     = f + o; o += N * HEADS;
  w.ex        = f + o; o += E * HEADS;
  w.bnsum     = f + o; o += 512;

  // ---- degrees & symmetric-norm scalars ----
  fill_kernel<<<256, 256, 0, stream>>>(w.out_isqrt, (long long)N, 0.0f);
  fill_kernel<<<256, 256, 0, stream>>>(w.in_sqrt,   (long long)N, 0.0f);
  degree_kernel<<<cdiv((int)E, 256), 256, 0, stream>>>(src, dst, w.out_isqrt, w.in_sqrt, (int)E);
  degnorm_kernel<<<cdiv((int)N, 256), 256, 0, stream>>>(w.out_isqrt, w.in_sqrt, (int)N);

  // ---- layer 0: GATConv 256 -> 4x64, write h (N,256) ----
  gat_conv(stream, w, x, IN_FEATS, HEADS, HID, W0, al0, ar0, resW0, src, dst,
           w.h, WFULL, 0, /*addSelf=*/0, /*bias=*/nullptr);

  // ---- reversible middle layers: group additive coupling, GROUP=2 ----
  for (int l = 0; l < 2; ++l) {
    for (int g = 0; g < 2; ++g) {
      // g==0: z = BN(x1)=h[:,128:], conv -> add into h[:, :128] (y0)
      // g==1: z = BN(y0)=h[:, :128], conv -> add into h[:,128:] (y1)
      int in_off  = (g == 0) ? GW : 0;
      int out_off = (g == 0) ? 0  : GW;
      int idx = l * 2 + g;
      bn_relu(stream, w, w.h, WFULL, in_off, GW,
              bn_gamma + (size_t)idx * GW, bn_beta + (size_t)idx * GW, w.z);
      gat_conv(stream, w, w.z, GW, HEADS, GW / HEADS,
               mid_W    + (size_t)idx * GW * GW,
               mid_al   + (size_t)idx * GW,
               mid_ar   + (size_t)idx * GW,
               mid_resW + (size_t)idx * GW * GW,
               src, dst, w.h, WFULL, out_off, /*addSelf=*/1, /*bias=*/nullptr);
    }
  }

  // ---- final norm + relu, then output GATConv 256 -> 1x40 (+bias) ----
  bn_relu(stream, w, w.h, WFULL, 0, WFULL, norm_g, norm_b, w.z);
  gat_conv(stream, w, w.z, WFULL, 1, CLASSES, W_last, al_last, ar_last, resW_last,
           src, dst, out, CLASSES, 0, /*addSelf=*/0, bias_last);
}
